// IMITERSelfAttention_644245094822
// MI455X (gfx1250) — matrix operand order repaired and checked by reading, not compiled, not run on hardware
//
#include <hip/hip_runtime.h>
#include <hip/hip_bf16.h>

#define DI __device__ __forceinline__

typedef __bf16 bf16_t;
typedef __attribute__((ext_vector_type(16))) __bf16 v16bf;
typedef __attribute__((ext_vector_type(8)))  __bf16 v8bf;
typedef __attribute__((ext_vector_type(8)))  float  v8f;
typedef __attribute__((ext_vector_type(4)))  unsigned int v4u;
typedef __attribute__((ext_vector_type(8)))  int v8i;
typedef __attribute__((ext_vector_type(4)))  int v4i;

// Problem constants
constexpr int B_ = 16, S_ = 1089, E_ = 1024, H_ = 16, D_ = 64, T_ = 512, I_ = 577;
constexpr int SP = 1152;   // padded S, multiple of 32, == T_ + IP
constexpr int IP = 640;    // padded I, multiple of 64
constexpr int TP = 512;    // T already aligned

// ---------------------------------------------------------------------------
// WMMA fragment loaders (CDNA5 wave32 layouts, CDNA5 ISA §7.12.2)
// ---------------------------------------------------------------------------
DI v16bf load_a_frag(const bf16_t* A, int lda, int m0, int k0) {
  const int lane = threadIdx.x & 31;
  const int row  = lane & 15;
  const int kb   = (lane >> 4) * 8;
  const bf16_t* p = A + (size_t)(m0 + row) * lda + (k0 + kb);
  v8bf c0 = *(const v8bf*)p;
  v8bf c1 = *(const v8bf*)(p + 16);
  v16bf f;
#pragma unroll
  for (int i = 0; i < 8; ++i) { f[i] = c0[i]; f[i + 8] = c1[i]; }
  return f;
}

DI v16bf load_b_frag(const bf16_t* Bt, int ldb, int n0, int k0) {
  const int lane = threadIdx.x & 31;
  const int col  = lane & 15;
  const int kb   = (lane >> 4) * 8;
  const bf16_t* p = Bt + (size_t)(n0 + col) * ldb + (k0 + kb);
  v8bf c0 = *(const v8bf*)p;
  v8bf c1 = *(const v8bf*)(p + 16);
  v16bf f;
#pragma unroll
  for (int i = 0; i < 8; ++i) { f[i] = c0[i]; f[i + 8] = c1[i]; }
  return f;
}

DI v8f wmma_bf16(v16bf a, v16bf b, v8f c) {
  return __builtin_amdgcn_wmma_f32_16x16x32_bf16(false, a, false, b, (short)0, c,
                                                 false, false);
}

// 32x64 tile per wave, B direct from global (B^T = N x K row-major).
DI void gemm_core_32x64(const bf16_t* __restrict__ A, int lda, int m0,
                        const bf16_t* __restrict__ Bt, int ldb, int n0,
                        int ksteps, v8f acc[2][4]) {
  for (int kk = 0; kk < ksteps; ++kk) {
    const int k0 = kk * 32;
    v16bf a0 = load_a_frag(A, lda, m0, k0);
    v16bf a1 = load_a_frag(A, lda, m0 + 16, k0);
#pragma unroll
    for (int t = 0; t < 4; ++t) {
      v16bf b = load_b_frag(Bt, ldb, n0 + t * 16, k0);
      acc[0][t] = wmma_bf16(a0, b, acc[0][t]);
      acc[1][t] = wmma_bf16(a1, b, acc[1][t]);
    }
  }
}

// ---------------------------------------------------------------------------
// Tensor Data Mover: DMA a 2D bf16 tile (tile_x cols, tile_y rows, row stride
// stride_elems) from global into LDS at byte offset lds_addr.
// D# packing per CDNA5 ISA (count=1, data_size=2B,
// type=2, no padding/iteration/multicast).
// ---------------------------------------------------------------------------
DI void tdm_load_2d(unsigned lds_addr, const void* gptr,
                    unsigned tile_x, unsigned tile_y, unsigned stride_elems) {
  unsigned long long ga = (unsigned long long)(uintptr_t)gptr;
  v4u g0;
  g0[0] = 1u;                                   // count=1 user descriptor
  g0[1] = lds_addr;                             // LDS byte address
  g0[2] = (unsigned)ga;                         // global_addr[31:0]
  g0[3] = ((unsigned)(ga >> 32) & 0x01FFFFFFu)  // global_addr[56:32]
          | (2u << 30);                         // type = 2 ("image")
  v8i g1;
  g1[0] = (int)(1u << 16);                      // data_size=1 -> 2 bytes
  g1[1] = (int)((tile_x & 0xFFFFu) << 16);      // tensor_dim0[15:0]
  g1[2] = (int)((tile_y & 0xFFFFu) << 16);      // tensor_dim1[15:0]
  g1[3] = (int)((tile_x & 0xFFFFu) << 16);      // tile_dim0
  g1[4] = (int)(tile_y & 0xFFFFu);              // tile_dim1 (tile_dim2=0)
  g1[5] = (int)stride_elems;                    // tensor_dim0_stride[31:0]
  g1[6] = 0;
  g1[7] = 0;
  v4i z4 = {0, 0, 0, 0};
#if defined(__clang_major__) && __clang_major__ >= 23
  v8i z8 = {0, 0, 0, 0, 0, 0, 0, 0};
  __builtin_amdgcn_tensor_load_to_lds(g0, g1, z4, z4, z8, 0);
#else
  __builtin_amdgcn_tensor_load_to_lds(g0, g1, z4, z4, 0);
#endif
}

// ---------------------------------------------------------------------------
// fp32 -> bf16 converters (with zero padding)
// ---------------------------------------------------------------------------
__global__ void conv_hidden_kernel(const float* __restrict__ in, bf16_t* __restrict__ out) {
  const long long n = (long long)B_ * SP * E_;
  for (long long idx = (long long)blockIdx.x * blockDim.x + threadIdx.x; idx < n;
       idx += (long long)gridDim.x * blockDim.x) {
    int e = (int)(idx % E_);
    long long t = idx / E_;
    int s = (int)(t % SP);
    int b = (int)(t / SP);
    float v = (s < S_) ? in[((long long)b * S_ + s) * E_ + e] : 0.f;
    out[idx] = (bf16_t)v;
  }
}

__global__ void conv_w_kernel(const float* __restrict__ in, bf16_t* __restrict__ out,
                              int R, int C, int Rp, int Cp) {
  const long long n = (long long)Rp * Cp;
  for (long long idx = (long long)blockIdx.x * blockDim.x + threadIdx.x; idx < n;
       idx += (long long)gridDim.x * blockDim.x) {
    int c = (int)(idx % Cp);
    int r = (int)(idx / Cp);
    float v = (r < R && c < C) ? in[(long long)r * C + c] : 0.f;
    out[idx] = (bf16_t)v;
  }
}

__global__ void zero_loss_kernel(float* p) {
  if (blockIdx.x == 0 && threadIdx.x == 0) *p = 0.f;
}

// ---------------------------------------------------------------------------
// QKV projection GEMM, 32x64 tile per wave.
// B (weight) tiles streamed by the Tensor Data Mover into double-buffered LDS
// (depth-1 prefetch, branch-free steady state, peeled tail); A fragments come
// through the VMEM pipe.
// transposed==0 -> q slab (S x D) row-major ; ==1 -> kT/vT slab (D x S)
// ---------------------------------------------------------------------------
__global__ __launch_bounds__(32, 1)
void qkv_gemm_kernel(const bf16_t* __restrict__ Abf,  // (B, SP, E)
                     const bf16_t* __restrict__ Wbf,  // (E, E) == B^T
                     const float* __restrict__ bias,  // (E)
                     bf16_t* __restrict__ out, int transposed) {
  __shared__ bf16_t Bl[2][64 * 32];  // two 4KB B tiles (64 rows n x 32 cols k)

  const int b  = blockIdx.z;
  const int m0 = blockIdx.y * 32;   // s
  const int n0 = blockIdx.x * 64;   // e_out (one head per tile group)
  const bf16_t* A  = Abf + (size_t)b * SP * E_;
  const bf16_t* Bt = Wbf + (size_t)n0 * E_;  // rows n0..n0+63 of B^T

  const unsigned lds0 = (unsigned)(uintptr_t)&Bl[0][0];
  const unsigned lds1 = (unsigned)(uintptr_t)&Bl[1][0];
  const bf16_t* Bl0 = &Bl[0][0];
  const bf16_t* Bl1 = &Bl[1][0];

  constexpr int KSTEPS = E_ / 32;
  // prime buffer 0 with tile 0
  tdm_load_2d(lds0, Bt, 32, 64, E_);

  v8f acc[2][4];
#pragma unroll
  for (int g = 0; g < 2; ++g)
#pragma unroll
    for (int t = 0; t < 4; ++t)
#pragma unroll
      for (int i = 0; i < 8; ++i) acc[g][t][i] = 0.f;

#pragma unroll 1
  for (int kk = 0; kk < KSTEPS - 1; ++kk) {
    const int k0 = kk * 32;
    // all ds reads of the buffer we are about to overwrite are done
    asm volatile("s_wait_dscnt 0" ::: "memory");
    // tile kk+1 -> buffer (kk+1)&1
    tdm_load_2d((kk & 1) ? lds0 : lds1, Bt + (k0 + 32), 32, 64, E_);
    __builtin_amdgcn_s_wait_tensorcnt(1);  // tile kk has landed
    asm volatile("" ::: "memory");

    v16bf a0 = load_a_frag(A, E_, m0, k0);
    v16bf a1 = load_a_frag(A, E_, m0 + 16, k0);
    const bf16_t* Bcur = (kk & 1) ? Bl1 : Bl0;
#pragma unroll
    for (int t = 0; t < 4; ++t) {
      v16bf bfr = load_b_frag(Bcur, 32, t * 16, 0);
      acc[0][t] = wmma_bf16(a0, bfr, acc[0][t]);
      acc[1][t] = wmma_bf16(a1, bfr, acc[1][t]);
    }
  }
  {  // peeled last k-step (tile KSTEPS-1 lives in buffer 1: KSTEPS is even)
    const int k0 = (KSTEPS - 1) * 32;
    __builtin_amdgcn_s_wait_tensorcnt(0);
    asm volatile("" ::: "memory");
    v16bf a0 = load_a_frag(A, E_, m0, k0);
    v16bf a1 = load_a_frag(A, E_, m0 + 16, k0);
#pragma unroll
    for (int t = 0; t < 4; ++t) {
      v16bf bfr = load_b_frag(Bl1, 32, t * 16, 0);
      acc[0][t] = wmma_bf16(a0, bfr, acc[0][t]);
      acc[1][t] = wmma_bf16(a1, bfr, acc[1][t]);
    }
  }

  const int lane = threadIdx.x & 31;
  const int half = lane >> 4, cl = lane & 15;
  const int h = n0 >> 6;
  const size_t slab = (size_t)b * H_ + h;
#pragma unroll
  for (int g = 0; g < 2; ++g)
#pragma unroll
    for (int t = 0; t < 4; ++t)
#pragma unroll
      for (int r = 0; r < 8; ++r) {
        int m = m0 + g * 16 + r + half * 8;  // s (< SP always)
        int n = n0 + t * 16 + cl;            // e_out
        int d = n & 63;
        float v = acc[g][t][r] + bias[n];
        bf16_t bv = (m < S_) ? (bf16_t)v : (bf16_t)0.f;
        if (!transposed) out[(slab * SP + m) * D_ + d] = bv;
        else             out[(slab * D_ + d) * SP + m] = bv;
      }
}

// ---------------------------------------------------------------------------
// ik imitation GEMM: ik[row_off+m, d] = sum_k Wa[m,k] * kT[d, coloff+k] + bias[m]
// 32x64 tile per wave (grid.x == 1, N = D = 64)
// ---------------------------------------------------------------------------
__global__ __launch_bounds__(32, 1)
void ik_gemm_kernel(const bf16_t* __restrict__ Wa, int lda,
                    const float* __restrict__ bias,
                    const bf16_t* __restrict__ kT, int coloff,
                    int ksteps, int Mv, int row_off,
                    bf16_t* __restrict__ ik) {
  const int slab = blockIdx.z;
  const int m0   = blockIdx.y * 32;
  const bf16_t* Bt = kT + (size_t)slab * D_ * SP + coloff;  // ldb = SP, rows = d
  v8f acc[2][4];
#pragma unroll
  for (int g = 0; g < 2; ++g)
#pragma unroll
    for (int t = 0; t < 4; ++t)
#pragma unroll
      for (int i = 0; i < 8; ++i) acc[g][t][i] = 0.f;
  gemm_core_32x64(Wa, lda, m0, Bt, SP, 0, ksteps, acc);

  const int lane = threadIdx.x & 31;
  const int half = lane >> 4, cl = lane & 15;
#pragma unroll
  for (int g = 0; g < 2; ++g)
#pragma unroll
    for (int t = 0; t < 4; ++t)
#pragma unroll
      for (int r = 0; r < 8; ++r) {
        int m = m0 + g * 16 + r + half * 8;
        int d = t * 16 + cl;
        bf16_t bv = (m < Mv) ? (bf16_t)(acc[g][t][r] + bias[m]) : (bf16_t)0.f;
        ik[((size_t)slab * SP + row_off + m) * D_ + d] = bv;
      }
}

// ---------------------------------------------------------------------------
// ivT imitation GEMM (computed transposed):
// ivT[d, col_off+n] = sum_k vT[d, a_coloff+k] * Wt[n,k] + bias[n]
// ---------------------------------------------------------------------------
__global__ __launch_bounds__(32, 1)
void ivt_gemm_kernel(const bf16_t* __restrict__ vT, int a_coloff,
                     const bf16_t* __restrict__ Wt, int ldb,
                     const float* __restrict__ bias,
                     int ksteps, int Nv, int col_off,
                     bf16_t* __restrict__ ivT) {
  const int slab = blockIdx.z;
  const int m0   = blockIdx.y * 32;   // d
  const int n0   = blockIdx.x * 64;   // t or i
  const bf16_t* A = vT + (size_t)slab * D_ * SP + a_coloff;  // lda = SP
  v8f acc[2][4];
#pragma unroll
  for (int g = 0; g < 2; ++g)
#pragma unroll
    for (int t = 0; t < 4; ++t)
#pragma unroll
      for (int i = 0; i < 8; ++i) acc[g][t][i] = 0.f;
  gemm_core_32x64(A, SP, m0, Wt, ldb, n0, ksteps, acc);

  const int lane = threadIdx.x & 31;
  const int half = lane >> 4, cl = lane & 15;
#pragma unroll
  for (int g = 0; g < 2; ++g)
#pragma unroll
    for (int t = 0; t < 4; ++t)
#pragma unroll
      for (int r = 0; r < 8; ++r) {
        int d = m0 + g * 16 + r + half * 8;  // < 64 always
        int n = n0 + t * 16 + cl;
        bf16_t bv = (n < Nv) ? (bf16_t)(acc[g][t][r] + bias[n]) : (bf16_t)0.f;
        ivT[((size_t)slab * D_ + d) * SP + col_off + n] = bv;
      }
}

// ---------------------------------------------------------------------------
// Flash attention: one wave per 16-query tile per (b,h).
// scores = q @ ik^T * 1/8 ; online softmax ; O += P @ iv (via ivT)
// ---------------------------------------------------------------------------
__global__ __launch_bounds__(32, 1)
void attn_kernel(const bf16_t* __restrict__ q,
                 const bf16_t* __restrict__ ik,
                 const bf16_t* __restrict__ ivT,
                 float* __restrict__ ctx) {
  const int qb = blockIdx.x;
  const int h  = blockIdx.y;
  const int b  = blockIdx.z;
  const size_t slab = (size_t)b * H_ + h;
  const bf16_t* Qp = q   + slab * SP * D_;
  const bf16_t* Kp = ik  + slab * SP * D_;
  const bf16_t* Vt = ivT + slab * D_ * SP;

  __shared__ bf16_t Pl[16 * 32];

  const int lane = threadIdx.x & 31;
  const int half = lane >> 4, cl = lane & 15;
  const int m0 = qb * 16;
  const float scale = 0.125f;  // 1/sqrt(64)

  v16bf qa0 = load_a_frag(Qp, D_, m0, 0);
  v16bf qa1 = load_a_frag(Qp, D_, m0, 32);

  float mrun[8], lrun[8];
  v8f O[4];
#pragma unroll
  for (int t = 0; t < 4; ++t)
#pragma unroll
    for (int i = 0; i < 8; ++i) O[t][i] = 0.f;
#pragma unroll
  for (int r = 0; r < 8; ++r) { mrun[r] = -1e30f; lrun[r] = 0.f; }

  for (int j = 0; j < SP / 32; ++j) {
    const int kb0 = j * 32;
    v8f s0, s1;
#pragma unroll
    for (int i = 0; i < 8; ++i) { s0[i] = 0.f; s1[i] = 0.f; }
    {
      v16bf b00 = load_b_frag(Kp, D_, kb0, 0);
      v16bf b01 = load_b_frag(Kp, D_, kb0, 32);
      s0 = wmma_bf16(qa0, b00, s0);
      s0 = wmma_bf16(qa1, b01, s0);
      v16bf b10 = load_b_frag(Kp, D_, kb0 + 16, 0);
      v16bf b11 = load_b_frag(Kp, D_, kb0 + 16, 32);
      s1 = wmma_bf16(qa0, b10, s1);
      s1 = wmma_bf16(qa1, b11, s1);
    }
    // scale + mask padded keys
    const int c0 = kb0 + cl, c1 = kb0 + 16 + cl;
    float rmax[8];
#pragma unroll
    for (int r = 0; r < 8; ++r) {
      float a0 = s0[r] * scale; if (c0 >= S_) a0 = -1e30f;
      float a1 = s1[r] * scale; if (c1 >= S_) a1 = -1e30f;
      s0[r] = a0; s1[r] = a1;
      rmax[r] = fmaxf(a0, a1);
    }
#pragma unroll
    for (int off = 1; off < 16; off <<= 1)
#pragma unroll
      for (int r = 0; r < 8; ++r)
        rmax[r] = fmaxf(rmax[r], __shfl_xor(rmax[r], off, 32));

    float alpha[8], rsum[8];
#pragma unroll
    for (int r = 0; r < 8; ++r) {
      float mn = fmaxf(mrun[r], rmax[r]);
      alpha[r] = __expf(mrun[r] - mn);
      mrun[r] = mn;
      float p0 = __expf(s0[r] - mn);
      float p1 = __expf(s1[r] - mn);
      rsum[r] = p0 + p1;
      Pl[(r + half * 8) * 32 + cl] = (bf16_t)p0;
      Pl[(r + half * 8) * 32 + 16 + cl] = (bf16_t)p1;
    }
#pragma unroll
    for (int off = 1; off < 16; off <<= 1)
#pragma unroll
      for (int r = 0; r < 8; ++r)
        rsum[r] += __shfl_xor(rsum[r], off, 32);
#pragma unroll
    for (int r = 0; r < 8; ++r) lrun[r] = lrun[r] * alpha[r] + rsum[r];
#pragma unroll
    for (int t = 0; t < 4; ++t)
#pragma unroll
      for (int r = 0; r < 8; ++r) O[t][r] *= alpha[r];

    __syncthreads();
    v16bf pa = load_a_frag(Pl, 32, 0, 0);
#pragma unroll
    for (int t = 0; t < 4; ++t) {
      v16bf vb = load_b_frag(Vt, SP, t * 16, kb0);
      O[t] = wmma_bf16(pa, vb, O[t]);
    }
    __syncthreads();
  }

#pragma unroll
  for (int t = 0; t < 4; ++t)
#pragma unroll
    for (int r = 0; r < 8; ++r) {
      int s = m0 + r + half * 8;
      if (s < S_) {
        int d = t * 16 + cl;
        ctx[(((size_t)b * S_ + s) * H_ + h) * D_ + d] = O[t][r] / lrun[r];
      }
    }
}

// ---------------------------------------------------------------------------
// imitation loss: mean((ik-k)^2) + mean((iv-v)^2)
// ---------------------------------------------------------------------------
__global__ void loss_kernel(const bf16_t* __restrict__ kT, const bf16_t* __restrict__ ik,
                            const bf16_t* __restrict__ vT, const bf16_t* __restrict__ ivT,
                            float* __restrict__ loss_out) {
  const long long N = (long long)B_ * H_ * S_ * D_;
  float acc = 0.f;
  for (long long i = (long long)blockIdx.x * blockDim.x + threadIdx.x; i < N;
       i += (long long)gridDim.x * blockDim.x) {
    long long slab = i / ((long long)S_ * D_);
    int rem = (int)(i % ((long long)S_ * D_));
    int s = rem / D_, d = rem % D_;
    float kv  = (float)kT [(slab * D_ + d) * SP + s];
    float ikv = (float)ik [(slab * SP + s) * D_ + d];
    float vv  = (float)vT [(slab * D_ + d) * SP + s];
    float ivv = (float)ivT[(slab * D_ + d) * SP + s];
    float dk = ikv - kv, dv = ivv - vv;
    acc += dk * dk + dv * dv;
  }
  __shared__ float red[256];
  red[threadIdx.x] = acc;
  __syncthreads();
  for (int st = 128; st > 0; st >>= 1) {
    if ((int)threadIdx.x < st) red[threadIdx.x] += red[threadIdx.x + st];
    __syncthreads();
  }
  if (threadIdx.x == 0) atomicAdd(loss_out, red[0] * (1.0f / (float)N));
}

// ---------------------------------------------------------------------------
extern "C" void kernel_launch(void* const* d_in, const int* in_sizes, int n_in,
                              void* d_out, int out_size, void* d_ws, size_t ws_size,
                              hipStream_t stream) {
  (void)in_sizes; (void)n_in; (void)out_size; (void)ws_size;

  const float* hs    = (const float*)d_in[0];
  const float* Wq    = (const float*)d_in[1];
  const float* bq    = (const float*)d_in[2];
  const float* Wk    = (const float*)d_in[3];
  const float* bk    = (const float*)d_in[4];
  const float* Wv    = (const float*)d_in[5];
  const float* bv    = (const float*)d_in[6];
  const float* W_itk = (const float*)d_in[7];
  const float* b_itk = (const float*)d_in[8];
  const float* W_itv = (const float*)d_in[9];
  const float* b_itv = (const float*)d_in[10];
  const float* W_iik = (const float*)d_in[11];
  const float* b_iik = (const float*)d_in[12];
  const float* W_iiv = (const float*)d_in[13];
  const float* b_iiv = (const float*)d_in[14];

  float* ctx = (float*)d_out;
  float* loss = ctx + (size_t)B_ * S_ * H_ * D_;

  // workspace carve-up (bf16 elements, all offsets 256B aligned by construction)
  bf16_t* w = (bf16_t*)d_ws;
  size_t o = 0;
  const size_t HID  = o; o += (size_t)B_ * SP * E_;
  const size_t WQ   = o; o += (size_t)E_ * E_;
  const size_t WK   = o; o += (size_t)E_ * E_;
  const size_t WV   = o; o += (size_t)E_ * E_;
  const size_t WITK = o; o += (size_t)T_ * IP;
  const size_t WITV = o; o += (size_t)T_ * IP;
  const size_t WIIK = o; o += (size_t)IP * TP;
  const size_t WIIV = o; o += (size_t)IP * TP;
  const size_t SLAB = (size_t)B_ * H_ * SP * D_;
  const size_t QB   = o; o += SLAB;
  const size_t KT   = o; o += SLAB;
  const size_t VT   = o; o += SLAB;
  const size_t IKB  = o; o += SLAB;
  const size_t IVT  = o; o += SLAB;

  // 1) conversions (zero-padded bf16)
  conv_hidden_kernel<<<4096, 256, 0, stream>>>(hs, w + HID);
  conv_w_kernel<<<2048, 256, 0, stream>>>(Wq, w + WQ, E_, E_, E_, E_);
  conv_w_kernel<<<2048, 256, 0, stream>>>(Wk, w + WK, E_, E_, E_, E_);
  conv_w_kernel<<<2048, 256, 0, stream>>>(Wv, w + WV, E_, E_, E_, E_);
  conv_w_kernel<<<640, 256, 0, stream>>>(W_itk, w + WITK, T_, I_, T_, IP);
  conv_w_kernel<<<640, 256, 0, stream>>>(W_itv, w + WITV, T_, I_, T_, IP);
  conv_w_kernel<<<640, 256, 0, stream>>>(W_iik, w + WIIK, I_, T_, IP, TP);
  conv_w_kernel<<<640, 256, 0, stream>>>(W_iiv, w + WIIV, I_, T_, IP, TP);
  zero_loss_kernel<<<1, 32, 0, stream>>>(loss);

  // 2) QKV projections (TDM-staged B, 32x64 tiles)
  dim3 gqkv(E_ / 64, SP / 32, B_);
  qkv_gemm_kernel<<<gqkv, 32, 0, stream>>>(w + HID, w + WQ, bq, w + QB, 0);
  qkv_gemm_kernel<<<gqkv, 32, 0, stream>>>(w + HID, w + WK, bk, w + KT, 1);
  qkv_gemm_kernel<<<gqkv, 32, 0, stream>>>(w + HID, w + WV, bv, w + VT, 1);

  // 3) imitation GEMMs (32x64 tiles)
  ik_gemm_kernel<<<dim3(1, T_ / 32, B_ * H_), 32, 0, stream>>>(
      w + WITK, IP, b_itk, w + KT, T_, IP / 32, T_, 0, w + IKB);          // pred_text
  ik_gemm_kernel<<<dim3(1, IP / 32, B_ * H_), 32, 0, stream>>>(
      w + WIIK, TP, b_iik, w + KT, 0, TP / 32, I_, T_, w + IKB);          // pred_image
  ivt_gemm_kernel<<<dim3(T_ / 64, D_ / 32, B_ * H_), 32, 0, stream>>>(
      w + VT, T_, w + WITV, IP, b_itv, IP / 32, T_, 0, w + IVT);          // pred_text
  ivt_gemm_kernel<<<dim3(IP / 64, D_ / 32, B_ * H_), 32, 0, stream>>>(
      w + VT, 0, w + WIIV, TP, b_iiv, TP / 32, I_, T_, w + IVT);          // pred_image

  // 4) flash attention
  attn_kernel<<<dim3(SP / 16, H_, B_), 32, 0, stream>>>(w + QB, w + IKB, w + IVT, ctx);

  // 5) imitation loss
  loss_kernel<<<2048, 256, 0, stream>>>(w + KT, w + IKB, w + VT, w + IVT, loss);
}
